// TFMambaBlock_6854767805184
// MI455X (gfx1250) — matrix operand order repaired and checked by reading, not compiled, not run on hardware
//
#include <hip/hip_runtime.h>
#include <hip/hip_bf16.h>

// ---------------------------------------------------------------------------
// TF-Mamba block for MI455X (gfx1250, wave32).
// GEMMs run on v_wmma_f32_16x16x32_bf16 (bf16 A/B built per ISA 7.12.2
// fragment layouts, f32 accumulate).  A-tiles are staged into LDS with
// GLOBAL_LOAD_ASYNC_TO_LDS_B128 (ASYNCcnt-tracked, VGPR-bypassing).  The
// selective scan runs one wave per sequence with the 128x16 state held in
// registers (64 f32/lane) and exp(dt*A) as exp2f -> v_exp_f32.
// Workspace layout (floats), NL = 32768 rows in both stages:
//   A_xz (NL*256) | Bc (NL*128) | Cd (NL*48) | Dt (NL*128) | Ey (NL*128)
//   Fc (NL*128)   | Gu (NL*64)  | Hu (NL*64) | Ur (NL*64)   => ~132 MB
// ---------------------------------------------------------------------------

typedef __attribute__((ext_vector_type(16))) __bf16 v16bf;
typedef __attribute__((ext_vector_type(8)))  float  v8f;

#define NL_ROWS 32768

// LDS byte-offset of a pointer that is known to live in LDS (addrspace(3)
// pointers are 32-bit LDS offsets on amdgcn -- exactly what the VDST operand
// of global_load_async_to_lds expects).
__device__ static inline unsigned lds_offset(const void* p) {
  return (unsigned)(unsigned long long)
      (__attribute__((address_space(3))) const char*)p;
}

// ---------------------------------------------------------------------------
// Generic WMMA GEMM:  D[M x Nw] = A[M x K] * W^T   (W is [Nw x K] row-major,
// or [K x Nw] when transW).  K is a template parameter so the k-loop fully
// unrolls (K is 64 or 128 everywhere in this pipeline).  Epilogue: optional
// bias[Nw], optional residual res[dstRow*ldr + n], optional per-sequence
// time flip of the destination row, destination column offset (concat).
// Block = 128 threads = 4 waves; each wave owns a 16x16 tile, block covers
// 64 rows x 16 cols.  A tile copied global->LDS with async-to-LDS (branch-
// free, 4 x b128 per thread); W tile staged with guarded scalar loads
// (needs zero-fill past Nw).
// ---------------------------------------------------------------------------
template <int K>
__global__ __launch_bounds__(128) void gemm_wmma_kernel(
    const float* __restrict__ A, int lda,
    const float* __restrict__ W, int ldw, int transW,
    float* __restrict__ D, int ldd, int col_off,
    const float* __restrict__ res, int ldr,
    const float* __restrict__ bias,
    int Nw, int flipL, int L)
{
  __shared__ float sA[64 * 32];
  __shared__ float sW[16 * 32];

  const int tid   = threadIdx.x;
  const int wave  = tid >> 5;
  const int lane  = tid & 31;
  const int half  = lane >> 4;
  const int mrow  = lane & 15;
  const int mBase = blockIdx.x * 64;
  const int nTile = blockIdx.y * 16;

  v8f acc = {};

#pragma unroll
  for (int k0 = 0; k0 < K; k0 += 32) {
    // ---- A tile (64 x 32 f32): async copy global -> LDS, 16B per lane op,
    // straight-line (constant trip count => no exec-mask branching around
    // the async issues).  Rows always in range (M multiple of 64), K a
    // multiple of 32 for every GEMM in this pipeline.
#pragma unroll
    for (int it = 0; it < 4; ++it) {
      int i  = tid + it * 128;    // 0..511
      int r  = i >> 3;            // row 0..63
      int c4 = (i & 7) << 2;      // col 0,4,..,28
      const float* gp = A + (long)(mBase + r) * lda + k0 + c4;
      unsigned loff = lds_offset(&sA[r * 32 + c4]);
      unsigned long long ga = (unsigned long long)gp;
      asm volatile("global_load_async_to_lds_b128 %0, %1, off"
                   :: "v"(loff), "v"(ga) : "memory");
    }
    // ---- W tile (16 x 32) with zero-fill past Nw (guarded scalar loads;
    // async cannot mask, and unguarded reads would run past x_proj_w).
#pragma unroll
    for (int it = 0; it < 4; ++it) {
      int i  = tid + it * 128;    // 0..511
      int n  = i >> 5;
      int c  = i & 31;
      int gn = nTile + n;
      int gk = k0 + c;
      float v = 0.0f;
      if (gn < Nw) v = transW ? W[(long)gk * ldw + gn] : W[(long)gn * ldw + gk];
      sW[i] = v;
    }
    // Each wave drains its own async queue, then block-wide barrier makes
    // the LDS tile visible to all waves.
    asm volatile("s_wait_asynccnt 0x0" ::: "memory");
    __syncthreads();

    // Build fragments per ISA 7.12.2 (16-bit A 16x32, B 32x16) and issue WMMA.
    v16bf afrag, bfrag;
    const float* aRow = sA + (wave * 16 + mrow) * 32;
#pragma unroll
    for (int e = 0; e < 16; ++e) {
      int ka = ((e >> 3) << 4) + (half << 3) + (e & 7);   // A: K index
      afrag[e] = (__bf16)aRow[ka];
      int kb = ((e >> 3) << 4) + (half << 3) + (e & 7);   // B: K index
      bfrag[e] = (__bf16)sW[mrow * 32 + kb];              // mrow = N column
    }
    acc = __builtin_amdgcn_wmma_f32_16x16x32_bf16(
        false, afrag, false, bfrag, (short)0, acc, false, false);
    __syncthreads();
  }

  // Store: VGPR r -> M = r + half*8, N = lane&15.
  const int gn = nTile + mrow;
  if (gn < Nw) {
#pragma unroll
    for (int r = 0; r < 8; ++r) {
      int gm = mBase + wave * 16 + half * 8 + r;
      int dstRow = gm;
      if (flipL) {
        int seq = gm / L;
        int l   = gm - seq * L;
        dstRow  = seq * L + (L - 1 - l);
      }
      float v = acc[r];
      if (bias) v += bias[gn];
      if (res)  v += res[(long)dstRow * ldr + gn];
      D[(long)dstRow * ldd + col_off + gn] = v;
    }
  }
}

// ---------------------------------------------------------------------------
// RMSNorm over the 64-channel axis; one wave per row, 2 channels/lane,
// butterfly reduce.
// ---------------------------------------------------------------------------
__global__ __launch_bounds__(256) void rmsnorm_kernel(
    const float* __restrict__ u, const float* __restrict__ w,
    float* __restrict__ out)
{
  int row  = blockIdx.x * 8 + (threadIdx.x >> 5);
  int lane = threadIdx.x & 31;
  const float* ur = u + (long)row * 64;
  float2 v = *(const float2*)(ur + lane * 2);
  float ss = v.x * v.x + v.y * v.y;
#pragma unroll
  for (int off = 16; off > 0; off >>= 1) ss += __shfl_xor(ss, off, 32);
  float scale = rsqrtf(ss * (1.0f / 64.0f) + 1e-5f);
  out[(long)row * 64 + lane * 2 + 0] = v.x * scale * w[lane * 2 + 0];
  out[(long)row * 64 + lane * 2 + 1] = v.y * scale * w[lane * 2 + 1];
}

// ---------------------------------------------------------------------------
// Causal depthwise conv (kernel 4) over L, + bias + SiLU.
// Input: xz rows of 256 (x = cols 0..127).  Output: Bc rows of 128.
// ---------------------------------------------------------------------------
__global__ __launch_bounds__(256) void conv_silu_kernel(
    const float* __restrict__ xz, const float* __restrict__ cw,
    const float* __restrict__ cb, float* __restrict__ out, int Ll)
{
  int idx = blockIdx.x * 256 + threadIdx.x;   // over NL*128
  int d = idx & 127;
  int m = idx >> 7;
  int l = m % Ll;
  int n = m / Ll;
  const float* base = xz + (long)n * Ll * 256 + d;
  float acc = cb[d];
#pragma unroll
  for (int j = 0; j < 4; ++j) {
    int t = l - 3 + j;
    if (t >= 0) acc += cw[d * 4 + j] * base[(long)t * 256];
  }
  out[idx] = acc / (1.0f + __expf(-acc));     // SiLU
}

// ---------------------------------------------------------------------------
// dt = softplus(xdbl[:, :4] @ dt_proj_w^T + dt_proj_b)
// ---------------------------------------------------------------------------
__global__ __launch_bounds__(256) void dt_kernel(
    const float* __restrict__ xdbl, const float* __restrict__ w,
    const float* __restrict__ b, float* __restrict__ dt)
{
  int idx = blockIdx.x * 256 + threadIdx.x;   // over NL*128
  int d = idx & 127;
  int m = idx >> 7;
  const float* r = xdbl + (long)m * 48;
  float a = b[d];
#pragma unroll
  for (int j = 0; j < 4; ++j) a += r[j] * w[d * 4 + j];
  dt[idx] = (a > 20.0f) ? a : log1pf(__expf(a));
}

// ---------------------------------------------------------------------------
// Selective scan: one wave per sequence.  Lane owns 4 d_inner channels x 16
// states in registers.  Fused epilogue: y = (scan + x*Dp) * silu(z).
// ---------------------------------------------------------------------------
__global__ __launch_bounds__(32) void scan_kernel(
    const float* __restrict__ dt, const float* __restrict__ xc,
    const float* __restrict__ xdbl, const float* __restrict__ xz,
    const float* __restrict__ A_log, const float* __restrict__ Dp,
    float* __restrict__ y, int Ll)
{
  const int n    = blockIdx.x;
  const int lane = threadIdx.x;
  const int d0   = lane * 4;

  float A2[4][16];
  float Dv[4];
#pragma unroll
  for (int d = 0; d < 4; ++d) {
    Dv[d] = Dp[d0 + d];
#pragma unroll
    for (int s = 0; s < 16; ++s)
      A2[d][s] = -__expf(A_log[(d0 + d) * 16 + s]) * 1.44269504f;  // A*log2(e)
  }
  float h[4][16];
#pragma unroll
  for (int d = 0; d < 4; ++d)
#pragma unroll
    for (int s = 0; s < 16; ++s) h[d][s] = 0.0f;

  const long rowBase = (long)n * Ll;
  for (int l = 0; l < Ll; ++l) {
    const long m = rowBase + l;
    float4 dtv = *(const float4*)(dt + m * 128 + d0);
    float4 xv  = *(const float4*)(xc + m * 128 + d0);
    float4 zv  = *(const float4*)(xz + m * 256 + 128 + d0);
    const float* bc = xdbl + m * 48 + 4;     // B at cols 4..19, C at 20..35
    float Bv[16], Cv[16];
#pragma unroll
    for (int s4 = 0; s4 < 4; ++s4) {
      float4 tb = *(const float4*)(bc + s4 * 4);
      float4 tc = *(const float4*)(bc + 16 + s4 * 4);
      Bv[s4 * 4 + 0] = tb.x; Bv[s4 * 4 + 1] = tb.y;
      Bv[s4 * 4 + 2] = tb.z; Bv[s4 * 4 + 3] = tb.w;
      Cv[s4 * 4 + 0] = tc.x; Cv[s4 * 4 + 1] = tc.y;
      Cv[s4 * 4 + 2] = tc.z; Cv[s4 * 4 + 3] = tc.w;
    }
    float4 yo;
    float* yop = &yo.x;
    const float* dtp = &dtv.x;
    const float* xp  = &xv.x;
    const float* zp  = &zv.x;
#pragma unroll
    for (int d = 0; d < 4; ++d) {
      float dtd = dtp[d];
      float dx  = dtd * xp[d];
      float accy = 0.0f;
#pragma unroll
      for (int s = 0; s < 16; ++s) {
        float dA = exp2f(dtd * A2[d][s]);    // v_exp_f32
        h[d][s] = dA * h[d][s] + dx * Bv[s];
        accy += h[d][s] * Cv[s];
      }
      float zz = zp[d];
      yop[d] = (accy + xp[d] * Dv[d]) * (zz / (1.0f + __expf(-zz)));
    }
    *(float4*)(y + m * 128 + d0) = yo;
  }
}

// ---------------------------------------------------------------------------
// Layout shufflers.
// ---------------------------------------------------------------------------
__global__ __launch_bounds__(256) void ingest_kernel(
    const float* __restrict__ x, float* __restrict__ u)
{
  int idx = blockIdx.x * 256 + threadIdx.x;      // x linear (c, t, f)
  int f = idx & 127;
  int t = (idx >> 7) & 255;
  int c = idx >> 15;
  u[((long)f * 256 + t) * 64 + c] = x[idx];
}

__global__ __launch_bounds__(256) void egress_kernel(
    const float* __restrict__ u, float* __restrict__ out)
{
  int idx = blockIdx.x * 256 + threadIdx.x;      // u linear (t, f, c)
  int c = idx & 63;
  int f = (idx >> 6) & 127;
  int t = idx >> 13;
  out[((long)c * 256 + t) * 128 + f] = u[idx];
}

__global__ __launch_bounds__(256) void transpose01_kernel(
    const float* __restrict__ in, float* __restrict__ out, int D0, int D1)
{
  int idx = blockIdx.x * 256 + threadIdx.x;      // in linear (D0, D1, 64)
  int c = idx & 63;
  int j = (idx >> 6) % D1;
  int i = idx / (64 * D1);
  out[((long)j * D0 + i) * 64 + c] = in[idx];
}

__global__ __launch_bounds__(256) void flip_kernel(
    const float* __restrict__ in, float* __restrict__ out, int Ll)
{
  int idx = blockIdx.x * 256 + threadIdx.x;      // out linear (n, l, 64)
  int c = idx & 63;
  int m = idx >> 6;
  int l = m % Ll;
  int n = m / Ll;
  out[idx] = in[((long)n * Ll + (Ll - 1 - l)) * 64 + c];
}

// ---------------------------------------------------------------------------
// Host orchestration.
// ---------------------------------------------------------------------------
extern "C" void kernel_launch(void* const* d_in, const int* in_sizes, int n_in,
                              void* d_out, int out_size, void* d_ws, size_t ws_size,
                              hipStream_t stream)
{
  const float* x         = (const float*)d_in[0];
  const float* norm_w    = (const float*)d_in[1];
  const float* in_proj_w = (const float*)d_in[2];
  const float* conv_w    = (const float*)d_in[3];
  const float* conv_b    = (const float*)d_in[4];
  const float* x_proj_w  = (const float*)d_in[5];
  const float* dt_proj_w = (const float*)d_in[6];
  const float* dt_proj_b = (const float*)d_in[7];
  const float* A_log     = (const float*)d_in[8];
  const float* Dp        = (const float*)d_in[9];
  const float* out_proj_w= (const float*)d_in[10];
  const float* tlin_w    = (const float*)d_in[11];
  const float* tlin_b    = (const float*)d_in[12];
  const float* flin_w    = (const float*)d_in[13];
  const float* flin_b    = (const float*)d_in[14];
  float* out = (float*)d_out;

  const long NL = NL_ROWS;
  float* ws   = (float*)d_ws;
  float* A_xz = ws;                  // NL*256
  float* Bc   = A_xz + NL * 256;     // NL*128
  float* Cd   = Bc   + NL * 128;     // NL*48
  float* Dt   = Cd   + NL * 48;      // NL*128
  float* Ey   = Dt   + NL * 128;     // NL*128 (rmsnorm out, then scan y)
  float* Fc   = Ey   + NL * 128;     // NL*128 (concat(yf, yb))
  float* Gu   = Fc   + NL * 128;     // NL*64  (stage input u)
  float* Hu   = Gu   + NL * 64;      // NL*64  (stage output)
  float* Ur   = Hu   + NL * 64;      // NL*64  (time-reversed u)

  auto gemm = [&](const float* Am, int lda, const float* Wm, int ldw, int transW,
                  float* Dm, int ldd, int coff, const float* res, int ldr,
                  const float* bias, int Nw, int K, int flip, int L) {
    dim3 grid(NL_ROWS / 64, (Nw + 15) / 16);
    if (K == 64) {
      gemm_wmma_kernel<64><<<grid, 128, 0, stream>>>(
          Am, lda, Wm, ldw, transW, Dm, ldd, coff, res, ldr, bias, Nw, flip, L);
    } else {
      gemm_wmma_kernel<128><<<grid, 128, 0, stream>>>(
          Am, lda, Wm, ldw, transW, Dm, ldd, coff, res, ldr, bias, Nw, flip, L);
    }
  };

  auto run_mixer = [&](int m, const float* uin, int flip, int Nn, int Ll, int coff) {
    rmsnorm_kernel<<<NL_ROWS / 8, 256, 0, stream>>>(uin, norm_w + m * 64, Ey);
    gemm(Ey, 64, in_proj_w + (long)m * 256 * 64, 64, 0,
         A_xz, 256, 0, nullptr, 0, nullptr, 256, 64, 0, Ll);
    conv_silu_kernel<<<(NL * 128) / 256, 256, 0, stream>>>(
        A_xz, conv_w + m * 128 * 4, conv_b + m * 128, Bc, Ll);
    gemm(Bc, 128, x_proj_w + (long)m * 36 * 128, 128, 0,
         Cd, 48, 0, nullptr, 0, nullptr, 36, 128, 0, Ll);
    dt_kernel<<<(NL * 128) / 256, 256, 0, stream>>>(
        Cd, dt_proj_w + m * 128 * 4, dt_proj_b + m * 128, Dt);
    scan_kernel<<<Nn, 32, 0, stream>>>(
        Dt, Bc, Cd, A_xz, A_log + m * 128 * 16, Dp + m * 128, Ey, Ll);
    gemm(Ey, 128, out_proj_w + (long)m * 64 * 128, 128, 0,
         Fc, 128, coff, Gu, 64, nullptr, 64, 128, flip, Ll);
  };

  // u_t[f][t][c] = x[c][t][f]
  ingest_kernel<<<(64 * 256 * 128) / 256, 256, 0, stream>>>(x, Gu);

  // ---- time stage: N=128 sequences, L=256 ----
  run_mixer(0, Gu, /*flip=*/0, 128, 256, /*coff=*/0);
  flip_kernel<<<(NL * 64) / 256, 256, 0, stream>>>(Gu, Ur, 256);
  run_mixer(1, Ur, /*flip=*/1, 128, 256, /*coff=*/64);
  gemm(Fc, 128, tlin_w, 64, /*transW=*/1, Hu, 64, 0, Gu, 64, tlin_b,
       64, 128, 0, 256);
  // (f,t,c) -> (t,f,c)
  transpose01_kernel<<<(NL * 64) / 256, 256, 0, stream>>>(Hu, Gu, 128, 256);

  // ---- frequency stage: N=256 sequences, L=128 ----
  run_mixer(2, Gu, 0, 256, 128, 0);
  flip_kernel<<<(NL * 64) / 256, 256, 0, stream>>>(Gu, Ur, 128);
  run_mixer(3, Ur, 1, 256, 128, 64);
  gemm(Fc, 128, flin_w, 64, 1, Hu, 64, 0, Gu, 64, flin_b, 64, 128, 0, 128);

  // (t,f,c) -> (c,t,f)
  egress_kernel<<<(NL * 64) / 256, 256, 0, stream>>>(Hu, out);
}
